// MultiHeadSelfAttention_84507776516504
// MI455X (gfx1250) — matrix operand order repaired and checked by reading, not compiled, not run on hardware
//
#include <hip/hip_runtime.h>

#define EMBED 1024
#define NH 16
#define DH 64
#define BATCH 2
#define SEQ 2048
#define MROWS (BATCH * SEQ)
#define GK EMBED  // K dimension of both GEMMs (compile-time)

typedef __attribute__((ext_vector_type(16))) _Float16 v16h;
typedef __attribute__((ext_vector_type(8)))  _Float16 v8h;
typedef __attribute__((ext_vector_type(4)))  _Float16 v4h;
typedef __attribute__((ext_vector_type(8)))  float    v8f;
typedef int v4i_g __attribute__((vector_size(16)));  // matches builtin prototype

// ---------------------------------------------------------------------------
// CDNA5 async global->LDS staging (ASYNCcnt path), with manual fallback
#if __has_builtin(__builtin_amdgcn_global_load_async_to_lds_b128)
#define HAVE_ASYNC_LDS 1
#else
#define HAVE_ASYNC_LDS 0
#endif

__device__ __forceinline__ void stage16(_Float16* lds, const _Float16* g) {
#if HAVE_ASYNC_LDS
  __builtin_amdgcn_global_load_async_to_lds_b128((v4i_g*)g, (v4i_g*)lds, 0, 0);
#else
  *(uint4*)lds = *(const uint4*)g;
#endif
}

__device__ __forceinline__ void wait_async0() {
#if HAVE_ASYNC_LDS
#if __has_builtin(__builtin_amdgcn_s_wait_asynccnt)
  __builtin_amdgcn_s_wait_asynccnt(0);
#else
  asm volatile("s_wait_asynccnt 0" ::: "memory");
#endif
#endif
}

// ---------------------------------------------------------------------------
// Fragment loaders (CDNA5 ISA 7.12.2 layouts, wave32)
// A 16x32 f16: lane<16 -> M=lane, K chunks [hi*8,+8) and [16+hi*8,+8)
__device__ __forceinline__ v16h ld_frag_a(const _Float16* base, int stride, int lane) {
  const int mr = lane & 15;
  const int hi = (lane >> 4) & 1;
  const _Float16* p = base + mr * stride + hi * 8;
  v8h c0 = *(const v8h*)(p);
  v8h c1 = *(const v8h*)(p + 16);
  v16h r;
#pragma unroll
  for (int i = 0; i < 8; ++i) { r[i] = c0[i]; r[i + 8] = c1[i]; }
  return r;
}

// B 32x16 f16 (stored [n][k] in LDS): lane holds N=lane&15,
// K chunks [hi*8,+8) and [16+hi*8,+8) (same order as A).
__device__ __forceinline__ v16h ld_frag_b(const _Float16* base, int stride, int lane) {
  const int nr = lane & 15;
  const int hi = (lane >> 4) & 1;
  const _Float16* p = base + nr * stride + hi * 8;
  v8h c0 = *(const v8h*)(p);
  v8h c1 = *(const v8h*)(p + 16);
  v16h r;
#pragma unroll
  for (int i = 0; i < 8; ++i) { r[i] = c0[i]; r[i + 8] = c1[i]; }
  return r;
}

__device__ __forceinline__ v8f wmma_f16(v16h a, v16h b, v8f c) {
  return __builtin_amdgcn_wmma_f32_16x16x32_f16(false, a, false, b, (short)0, c,
                                                false, false);
}

// ---------------------------------------------------------------------------
// f32 -> f16 conversion
__global__ void cvt_f32_f16_kernel(const float4* __restrict__ src,
                                   v4h* __restrict__ dst, int n4) {
  int i = blockIdx.x * blockDim.x + threadIdx.x;
  if (i >= n4) return;
  float4 v = src[i];
  v4h h;
  h[0] = (_Float16)v.x; h[1] = (_Float16)v.y;
  h[2] = (_Float16)v.z; h[3] = (_Float16)v.w;
  dst[i] = h;
}

// ---------------------------------------------------------------------------
// Tiled WMMA GEMM: C[m,n] = sum_k A[m,k] * Bw[n,k] + bias[n], K = GK = 1024
// mode 0: scatter f16 into Q/K/V [B,H,T,Dh];  mode 1: dense f32 out
#define BM 128
#define BN 128
#define BK 32
#define LDA 40  // LDS row stride in halves (32 + 8 pad, keeps 16B alignment)

__device__ __forceinline__ void stage_tiles(_Float16* AsB, _Float16* BsB,
                                            const _Float16* A, const _Float16* Bw,
                                            int m0, int n0, int k0, int tid) {
  // A tile 128x32 and B tile 128x32: 512 x 16B chunks each, 2+2 per thread
#pragma unroll
  for (int c = 0; c < 2; ++c) {
    int idx = tid + c * 256;
    int row = idx >> 2, cc = idx & 3;
    stage16(AsB + row * LDA + cc * 8, A + (size_t)(m0 + row) * GK + k0 + cc * 8);
  }
#pragma unroll
  for (int c = 0; c < 2; ++c) {
    int idx = tid + c * 256;
    int row = idx >> 2, cc = idx & 3;
    stage16(BsB + row * LDA + cc * 8, Bw + (size_t)(n0 + row) * GK + k0 + cc * 8);
  }
}

__device__ __forceinline__ void compute_tile(const _Float16* AsB, const _Float16* BsB,
                                             v8f acc[2][4], int wm, int wn, int lane) {
  v16h af[2];
#pragma unroll
  for (int i = 0; i < 2; ++i)
    af[i] = ld_frag_a(AsB + (wm * 32 + i * 16) * LDA, LDA, lane);
#pragma unroll
  for (int j = 0; j < 4; ++j) {
    v16h bf = ld_frag_b(BsB + (wn * 64 + j * 16) * LDA, LDA, lane);
#pragma unroll
    for (int i = 0; i < 2; ++i) acc[i][j] = wmma_f16(af[i], bf, acc[i][j]);
  }
}

__global__ __launch_bounds__(256) void gemm_wmma_kernel(
    const _Float16* __restrict__ A, const _Float16* __restrict__ Bw,
    const float* __restrict__ bias,
    _Float16* __restrict__ Qd, _Float16* __restrict__ Kd, _Float16* __restrict__ Vd,
    float* __restrict__ Cf, int N, int mode) {
  __shared__ __align__(16) _Float16 As[2][BM * LDA];
  __shared__ __align__(16) _Float16 Bs[2][BN * LDA];

  const int tid = threadIdx.x;
  const int lane = tid & 31;
  const int wave = tid >> 5;     // 8 waves
  const int wm = wave >> 1;      // 0..3 (M): 32-row strip
  const int wn = wave & 1;       // 0..1 (N): 64-col strip
  const int m0 = blockIdx.y * BM;
  const int n0 = blockIdx.x * BN;

  v8f acc[2][4] = {};

  // Double-buffered software pipeline over K (async stage next || compute cur)
  stage_tiles(As[0], Bs[0], A, Bw, m0, n0, 0, tid);
  wait_async0();
  __syncthreads();
  int cur = 0;
  for (int k0 = 0; k0 < GK - BK; k0 += BK) {
    stage_tiles(As[cur ^ 1], Bs[cur ^ 1], A, Bw, m0, n0, k0 + BK, tid);
    compute_tile(As[cur], Bs[cur], acc, wm, wn, lane);
    wait_async0();
    __syncthreads();
    cur ^= 1;
  }
  compute_tile(As[cur], Bs[cur], acc, wm, wn, lane);

  const int hi = (lane >> 4) & 1;
  const int lc = lane & 15;
#pragma unroll
  for (int i = 0; i < 2; ++i) {
#pragma unroll
    for (int j = 0; j < 4; ++j) {
      const int col = n0 + wn * 64 + j * 16 + lc;
      const float bv = bias[col];
      const int rbase = m0 + wm * 32 + i * 16 + hi * 8;
#pragma unroll
      for (int r = 0; r < 8; ++r) {
        const int row = rbase + r;
        const float val = acc[i][j][r] + bv;
        if (mode == 0) {
          const int which = col >> 10;  // 0=Q,1=K,2=V
          const int rem = col & 1023;
          const int h = rem >> 6;
          const int dh = rem & 63;
          const int b = row >> 11;
          const int t = row & 2047;
          const size_t di = (((size_t)b * NH + h) * SEQ + t) * DH + dh;
          _Float16* dst = (which == 0) ? Qd : (which == 1) ? Kd : Vd;
          dst[di] = (_Float16)val;
        } else {
          Cf[(size_t)row * N + col] = val;
        }
      }
    }
  }
}

// ---------------------------------------------------------------------------
// Flash attention: block = 4 waves x 32, each wave owns 16 q-rows.
// K/V stream through double-buffered LDS tiles; online softmax in registers.
#define KSTR 72  // 64 + 8 pad halves
#define VSTR 40  // 32 + 8 pad halves

__device__ __forceinline__ void stage_kv(_Float16* KsB, _Float16* VsB,
                                         const _Float16* Kb, const _Float16* Vb,
                                         int kt0, int tid) {
  // K tile 32x64 -> KsB[kcol][dh] (straight copy, async-capable)
#pragma unroll
  for (int c = 0; c < 2; ++c) {
    int idx = tid + c * 128;
    int row = idx >> 3, cc = idx & 7;
    stage16(KsB + row * KSTR + cc * 8, Kb + (size_t)(kt0 + row) * DH + cc * 8);
  }
  // V tile 32x64 stored transposed -> VsB[dh][kcol] (manual: needs shuffle)
#pragma unroll
  for (int c = 0; c < 2; ++c) {
    int idx = tid + c * 128;
    int row = idx >> 3, cc = idx & 7;
    v8h v = *(const v8h*)(Vb + (size_t)(kt0 + row) * DH + cc * 8);
#pragma unroll
    for (int i = 0; i < 8; ++i) VsB[(cc * 8 + i) * VSTR + row] = v[i];
  }
}

__global__ __launch_bounds__(128) void flash_attn_kernel(
    const _Float16* __restrict__ Qg, const _Float16* __restrict__ Kg,
    const _Float16* __restrict__ Vg, _Float16* __restrict__ ctx) {
  __shared__ __align__(16) _Float16 Ks[2][32 * KSTR];
  __shared__ __align__(16) _Float16 Vs[2][64 * VSTR];
  __shared__ __align__(16) _Float16 Ps[4 * 16 * VSTR];  // per-wave P staging

  const int tid = threadIdx.x;
  const int lane = tid & 31;
  const int wave = tid >> 5;
  const int bh = blockIdx.y;  // b*NH + h
  const int q0 = blockIdx.x * 64 + wave * 16;
  const _Float16* Qb = Qg + (size_t)bh * SEQ * DH;
  const _Float16* Kb = Kg + (size_t)bh * SEQ * DH;
  const _Float16* Vb = Vg + (size_t)bh * SEQ * DH;

  const int mr = lane & 15;
  const int hi = (lane >> 4) & 1;

  // Q fragments (16x64 = two 16x32 A-frags) pinned in VGPRs for the whole loop
  v16h qf[2];
#pragma unroll
  for (int kc = 0; kc < 2; ++kc) {
    const _Float16* p = Qb + (size_t)(q0 + mr) * DH + kc * 32 + hi * 8;
    v8h c0 = *(const v8h*)p;
    v8h c1 = *(const v8h*)(p + 16);
#pragma unroll
    for (int i = 0; i < 8; ++i) { qf[kc][i] = c0[i]; qf[kc][i + 8] = c1[i]; }
  }

  v8f o[4] = {};
  float mrow[8], lrow[8];
#pragma unroll
  for (int r = 0; r < 8; ++r) { mrow[r] = -3.0e38f; lrow[r] = 0.0f; }

  const float scale = 0.125f;  // Dh^-0.5
  _Float16* pp = Ps + wave * 16 * VSTR;

  auto attn_step = [&](const _Float16* KsB, const _Float16* VsB) {
    // S = Q K^T : 16x32 tile as two 16x16 WMMAs, K-dim = Dh = 2x32
    v8f s[2] = {};
#pragma unroll
    for (int nt = 0; nt < 2; ++nt)
#pragma unroll
      for (int kc = 0; kc < 2; ++kc) {
        v16h bf = ld_frag_b(KsB + nt * 16 * KSTR + kc * 32, KSTR, lane);
        s[nt] = wmma_f16(qf[kc], bf, s[nt]);
      }

    // Online softmax; row r lives across 16 lanes of each half-wave
    float p0a[8], p1a[8];
#pragma unroll
    for (int r = 0; r < 8; ++r) {
      float a0 = s[0][r] * scale;
      float a1 = s[1][r] * scale;
      float mx = fmaxf(a0, a1);
#pragma unroll
      for (int d = 1; d < 16; d <<= 1) mx = fmaxf(mx, __shfl_xor(mx, d, 32));
      const float mn = fmaxf(mrow[r], mx);
      const float alpha = __expf(mrow[r] - mn);
      const float p0 = __expf(a0 - mn);
      const float p1 = __expf(a1 - mn);
      float rs = p0 + p1;
#pragma unroll
      for (int d = 1; d < 16; d <<= 1) rs += __shfl_xor(rs, d, 32);
      lrow[r] = lrow[r] * alpha + rs;
      mrow[r] = mn;
#pragma unroll
      for (int j = 0; j < 4; ++j) o[j][r] *= alpha;
      p0a[r] = p0;
      p1a[r] = p1;
    }

    // Restage P (C-layout f32) through per-wave LDS into A-frag f16 layout
#pragma unroll
    for (int r = 0; r < 8; ++r) {
      const int prow = r + hi * 8;
      pp[prow * VSTR + mr] = (_Float16)p0a[r];
      pp[prow * VSTR + 16 + mr] = (_Float16)p1a[r];
    }
    v16h pf = ld_frag_a(pp, VSTR, lane);

    // O += P @ V : 4 dh-tiles of 16, K-dim = 32
#pragma unroll
    for (int nt = 0; nt < 4; ++nt) {
      v16h vf = ld_frag_b(VsB + nt * 16 * VSTR, VSTR, lane);
      o[nt] = wmma_f16(pf, vf, o[nt]);
    }
  };

  // Double-buffered pipeline over the 64 K/V tiles
  stage_kv(Ks[0], Vs[0], Kb, Vb, 0, tid);
  wait_async0();
  __syncthreads();
  int cur = 0;
  for (int kt0 = 0; kt0 < SEQ - 32; kt0 += 32) {
    stage_kv(Ks[cur ^ 1], Vs[cur ^ 1], Kb, Vb, kt0 + 32, tid);
    attn_step(Ks[cur], Vs[cur]);
    wait_async0();
    __syncthreads();
    cur ^= 1;
  }
  attn_step(Ks[cur], Vs[cur]);

  // Normalize and write ctx in (B,T,D) layout, d = h*64 + dh (f16)
  const int b = bh >> 4;
  const int h = bh & 15;
#pragma unroll
  for (int nt = 0; nt < 4; ++nt) {
#pragma unroll
    for (int r = 0; r < 8; ++r) {
      const int t = q0 + r + hi * 8;
      const float val = o[nt][r] / lrow[r];
      const int dh = nt * 16 + mr;
      ctx[((size_t)b * SEQ + t) * EMBED + h * DH + dh] = (_Float16)val;
    }
  }
}

// ---------------------------------------------------------------------------
extern "C" void kernel_launch(void* const* d_in, const int* in_sizes, int n_in,
                              void* d_out, int out_size, void* d_ws, size_t ws_size,
                              hipStream_t stream) {
  (void)in_sizes; (void)n_in; (void)out_size; (void)ws_size;
  const float* x = (const float*)d_in[0];
  const float* qkv_w = (const float*)d_in[1];
  const float* qkv_b = (const float*)d_in[2];
  const float* out_w = (const float*)d_in[3];
  const float* out_b = (const float*)d_in[4];
  float* out = (float*)d_out;

  // Workspace layout (f16 elements). Total 48 MB.
  _Float16* ws = (_Float16*)d_ws;
  const size_t XH = 0;
  const size_t WQKV = XH + (size_t)MROWS * EMBED;
  const size_t OW = WQKV + (size_t)3 * EMBED * EMBED;
  const size_t QO = OW + (size_t)EMBED * EMBED;
  const size_t KO = QO + (size_t)BATCH * NH * SEQ * DH;
  const size_t VO = KO + (size_t)BATCH * NH * SEQ * DH;
  const size_t CX = VO + (size_t)BATCH * NH * SEQ * DH;

  _Float16* Xh = ws + XH;
  _Float16* Wq = ws + WQKV;
  _Float16* Ow = ws + OW;
  _Float16* Qt = ws + QO;
  _Float16* Kt = ws + KO;
  _Float16* Vt = ws + VO;
  _Float16* Cx = ws + CX;

  // 1) f32 -> f16
  cvt_f32_f16_kernel<<<(MROWS * EMBED / 4 + 255) / 256, 256, 0, stream>>>(
      (const float4*)x, (v4h*)Xh, MROWS * EMBED / 4);
  cvt_f32_f16_kernel<<<(3 * EMBED * EMBED / 4 + 255) / 256, 256, 0, stream>>>(
      (const float4*)qkv_w, (v4h*)Wq, 3 * EMBED * EMBED / 4);
  cvt_f32_f16_kernel<<<(EMBED * EMBED / 4 + 255) / 256, 256, 0, stream>>>(
      (const float4*)out_w, (v4h*)Ow, EMBED * EMBED / 4);

  // 2) QKV projection, scatter into per-head Q/K/V
  dim3 g1(3 * EMBED / BN, MROWS / BM);
  gemm_wmma_kernel<<<g1, 256, 0, stream>>>(Xh, Wq, qkv_b, Qt, Kt, Vt, nullptr,
                                           3 * EMBED, 0);

  // 3) Flash attention
  dim3 g2(SEQ / 64, BATCH * NH);
  flash_attn_kernel<<<g2, 128, 0, stream>>>(Qt, Kt, Vt, Cx);

  // 4) Output projection (f32 + bias)
  dim3 g3(EMBED / BN, MROWS / BM);
  gemm_wmma_kernel<<<g3, 256, 0, stream>>>(Cx, Ow, out_b, nullptr, nullptr,
                                           nullptr, out, EMBED, 1);
}
